// DeepSelfAttention_25348896981058
// MI455X (gfx1250) — matrix operand order repaired and checked by reading, not compiled, not run on hardware
//
#include <hip/hip_runtime.h>
#include <cstdint>
#include <cstddef>
#include <cmath>

// ---------------------------------------------------------------------------
// CDNA5 (gfx1250) wave32 WMMA types
// ---------------------------------------------------------------------------
typedef __attribute__((ext_vector_type(8)))  float   v8f;
typedef __attribute__((ext_vector_type(16))) __bf16  v16bf;
typedef __attribute__((ext_vector_type(8)))  __bf16  v8bf;
typedef __attribute__((ext_vector_type(4)))  __bf16  v4bf;
typedef __attribute__((ext_vector_type(4)))  int     v4i;

union FragBF { v16bf v; v8bf h[2]; };

#define LDS_STRIDE 40   // bf16 per LDS tile row: 32 data + 8 pad (80B, 16B-aligned)

// ---------------------------------------------------------------------------
// CDNA5 async global->LDS path (ASYNCcnt), with compile-time fallback.
// Builtin prototype (from hipcc diagnostic): (v4i AS1*, v4i AS3*, imm, imm)
// ---------------------------------------------------------------------------
#if defined(__AMDGCN__) && __has_builtin(__builtin_amdgcn_global_load_async_to_lds_b128)
#define USE_ASYNC_LDS 1
typedef __attribute__((address_space(1))) v4i as1_v4i;
typedef __attribute__((address_space(3))) v4i as3_v4i;
#else
#define USE_ASYNC_LDS 0
#endif

__device__ __forceinline__ void wait_asynccnt0() {
#if defined(__AMDGCN__)
#if __has_builtin(__builtin_amdgcn_s_wait_asynccnt)
  __builtin_amdgcn_s_wait_asynccnt(0);
#else
  asm volatile("s_wait_asynccnt 0x0" ::: "memory");
#endif
#endif
}

// ---------------------------------------------------------------------------
// fp32 -> bf16 conversion (vectorized x4)
// ---------------------------------------------------------------------------
__global__ __launch_bounds__(256) void cvt_f32_bf16(const float* __restrict__ in,
                                                    __bf16* __restrict__ out,
                                                    long n4) {
  long i = (long)blockIdx.x * blockDim.x + threadIdx.x;
  if (i >= n4) return;
  float4 v = reinterpret_cast<const float4*>(in)[i];
  v4bf o;
  o.x = (__bf16)v.x; o.y = (__bf16)v.y; o.z = (__bf16)v.z; o.w = (__bf16)v.w;
  reinterpret_cast<v4bf*>(out)[i] = o;
}

// ---------------------------------------------------------------------------
// Tiled bf16 GEMM, f32 WMMA accumulation, double-buffered LDS with async
// global->LDS copies:
//   C[m,n] = act( scale * sum_k A[m,k]*B[n,k] + bias[n] )   (i.e. C = A @ B^T)
// 128x128 block tile, BK=32, 8 waves in 2x4 grid, 64x32 per wave
// (4x2 fragments of v_wmma_f32_16x16x32_bf16).
// ---------------------------------------------------------------------------
__global__ __launch_bounds__(256) void gemm_bf16_wmma(
    const __bf16* __restrict__ A, const __bf16* __restrict__ B,
    const float* __restrict__ bias, __bf16* __restrict__ C,
    int M, int N, int K, int ldc, float scale, int relu, int transpose_out)
{
  __shared__ __bf16 ldsA[2][128 * LDS_STRIDE];
  __shared__ __bf16 ldsB[2][128 * LDS_STRIDE];

  const int t    = threadIdx.x;
  const int lane = t & 31;
  const int wave = t >> 5;
  const int wm   = wave >> 2;      // 0..1 : wave row   (64 rows each)
  const int wn   = wave & 3;       // 0..3 : wave col   (32 cols each)
  const int lm   = lane & 15;      // row/col inside fragment
  const int lh   = lane >> 4;      // lane half

  const int mBase = blockIdx.y * 128;
  const int nBase = blockIdx.x * 128;

  // cooperative tile-load mapping: thread -> (row, 16-elem k-chunk) = 32B
  const int lr = t >> 1;           // 0..127
  const int kq = (t & 1) * 16;     // 0 or 16

  const __bf16* gA = A + (size_t)(mBase + lr) * K + kq;
  const __bf16* gB = B + (size_t)(nBase + lr) * K + kq;
  const int ldsOff = lr * LDS_STRIDE + kq;   // element offset inside one tile

  const v8f vzero = {0.f, 0.f, 0.f, 0.f, 0.f, 0.f, 0.f, 0.f};
  v8f acc[4][2];
#pragma unroll
  for (int i = 0; i < 4; ++i)
#pragma unroll
    for (int j = 0; j < 2; ++j) acc[i][j] = vzero;

  // --- fragment loads (layouts per CDNA5 ISA 7.12.2) + 8 WMMAs -------------
  auto computeTile = [&](int b) {
    FragBF aF[4], bF[2];
#pragma unroll
    for (int i = 0; i < 4; ++i) {
      const char* base =
          (const char*)&ldsA[b][(size_t)(wm * 64 + i * 16 + lm) * LDS_STRIDE];
      aF[i].h[0] = *reinterpret_cast<const v8bf*>(base + lh * 16);
      aF[i].h[1] = *reinterpret_cast<const v8bf*>(base + 32 + lh * 16);
    }
#pragma unroll
    for (int j = 0; j < 2; ++j) {
      const char* base =
          (const char*)&ldsB[b][(size_t)(wn * 32 + j * 16 + lm) * LDS_STRIDE];
      bF[j].h[0] = *reinterpret_cast<const v8bf*>(base + lh * 16);
      bF[j].h[1] = *reinterpret_cast<const v8bf*>(base + 32 + lh * 16);
    }
#pragma unroll
    for (int i = 0; i < 4; ++i)
#pragma unroll
      for (int j = 0; j < 2; ++j)
        acc[i][j] = __builtin_amdgcn_wmma_f32_16x16x32_bf16(
            false, aF[i].v, false, bF[j].v, (short)0, acc[i][j], false, false);
  };

#if USE_ASYNC_LDS
  // -------- async pipeline: tile k+1 streams into buf^1 while buf computes
  auto issueAsync = [&](int k0, int b) {
    const __bf16* ga = gA + k0;
    const __bf16* gb = gB + k0;
    __bf16* la = &ldsA[b][ldsOff];
    __bf16* lb = &ldsB[b][ldsOff];
    __builtin_amdgcn_global_load_async_to_lds_b128((as1_v4i*)ga,
                                                   (as3_v4i*)la, 0, 0);
    __builtin_amdgcn_global_load_async_to_lds_b128((as1_v4i*)(ga + 8),
                                                   (as3_v4i*)(la + 8), 0, 0);
    __builtin_amdgcn_global_load_async_to_lds_b128((as1_v4i*)gb,
                                                   (as3_v4i*)lb, 0, 0);
    __builtin_amdgcn_global_load_async_to_lds_b128((as1_v4i*)(gb + 8),
                                                   (as3_v4i*)(lb + 8), 0, 0);
  };

  issueAsync(0, 0);
  wait_asynccnt0();
  __syncthreads();

  int buf = 0;
  for (int k0 = 0; k0 < K; k0 += 32) {
    const bool more = (k0 + 32) < K;
    if (more) issueAsync(k0 + 32, buf ^ 1);   // overlap with WMMA below
    computeTile(buf);
    if (more) {
      wait_asynccnt0();                        // my async writes landed
      __syncthreads();                         // everyone's landed
    }
    buf ^= 1;
  }
#else
  // -------- fallback: register-double-buffered synchronous pipeline
  uint4* lA0 = reinterpret_cast<uint4*>(&ldsA[0][ldsOff]);
  uint4* lB0 = reinterpret_cast<uint4*>(&ldsB[0][ldsOff]);
  uint4* lA1 = reinterpret_cast<uint4*>(&ldsA[1][ldsOff]);
  uint4* lB1 = reinterpret_cast<uint4*>(&ldsB[1][ldsOff]);

  {
    uint4 a0 = reinterpret_cast<const uint4*>(gA)[0];
    uint4 a1 = reinterpret_cast<const uint4*>(gA)[1];
    uint4 b0 = reinterpret_cast<const uint4*>(gB)[0];
    uint4 b1 = reinterpret_cast<const uint4*>(gB)[1];
    lA0[0] = a0; lA0[1] = a1; lB0[0] = b0; lB0[1] = b1;
  }
  __syncthreads();

  int buf = 0;
  for (int k0 = 0; k0 < K; k0 += 32) {
    const bool more = (k0 + 32) < K;
    uint4 na0, na1, nb0, nb1;
    if (more) {                                 // loads in flight during WMMAs
      na0 = reinterpret_cast<const uint4*>(gA + k0 + 32)[0];
      na1 = reinterpret_cast<const uint4*>(gA + k0 + 32)[1];
      nb0 = reinterpret_cast<const uint4*>(gB + k0 + 32)[0];
      nb1 = reinterpret_cast<const uint4*>(gB + k0 + 32)[1];
      __builtin_prefetch(gA + k0 + 64, 0, 1);
      __builtin_prefetch(gB + k0 + 64, 0, 1);
    }
    computeTile(buf);
    if (more) {
      __syncthreads();
      if (buf == 0) { lA1[0] = na0; lA1[1] = na1; lB1[0] = nb0; lB1[1] = nb1; }
      else          { lA0[0] = na0; lA0[1] = na1; lB0[0] = nb0; lB0[1] = nb1; }
      __syncthreads();
    }
    buf ^= 1;
  }
#endif

  // --- epilogue: scale, bias, ReLU, bf16 store (C layout: m = lh*8+r, n = lm)
#pragma unroll
  for (int i = 0; i < 4; ++i) {
#pragma unroll
    for (int j = 0; j < 2; ++j) {
      const int n = nBase + wn * 32 + j * 16 + lm;
      const float bval = bias ? bias[n] : 0.0f;
#pragma unroll
      for (int r = 0; r < 8; ++r) {
        const int m = mBase + wm * 64 + i * 16 + lh * 8 + r;
        float v = acc[i][j][r] * scale + bval;
        if (relu) v = fmaxf(v, 0.0f);
        const __bf16 o = (__bf16)v;
        if (!transpose_out) C[(size_t)m * ldc + n] = o;
        else                C[(size_t)n * ldc + m] = o;
      }
    }
  }
}

// ---------------------------------------------------------------------------
// Row softmax over bf16 matrix, in place. One workgroup per row.
// ---------------------------------------------------------------------------
__global__ __launch_bounds__(256) void softmax_rows(__bf16* __restrict__ S, int N) {
  __shared__ float red[256];
  const int row = blockIdx.x;
  const int t = threadIdx.x;
  __bf16* p = S + (size_t)row * N;

  float mx = -3.0e38f;
  for (int c = t; c < N; c += 256) mx = fmaxf(mx, (float)p[c]);
  red[t] = mx;
  __syncthreads();
  for (int s = 128; s > 0; s >>= 1) {
    if (t < s) red[t] = fmaxf(red[t], red[t + s]);
    __syncthreads();
  }
  mx = red[0];
  __syncthreads();

  float sum = 0.0f;
  for (int c = t; c < N; c += 256) sum += __expf((float)p[c] - mx);
  red[t] = sum;
  __syncthreads();
  for (int s = 128; s > 0; s >>= 1) {
    if (t < s) red[t] += red[t + s];
    __syncthreads();
  }
  const float inv = 1.0f / red[0];

  for (int c = t; c < N; c += 256)
    p[c] = (__bf16)(__expf((float)p[c] - mx) * inv);
}

// ---------------------------------------------------------------------------
// Final projection: out[m] = sum_k H[m,k] * w[k].  One wave32 per row.
// ---------------------------------------------------------------------------
__global__ __launch_bounds__(256) void final_matvec(const __bf16* __restrict__ H,
                                                    const float* __restrict__ w,
                                                    float* __restrict__ out, int D) {
  const int row  = blockIdx.x * 8 + (threadIdx.x >> 5);
  const int lane = threadIdx.x & 31;
  const __bf16* p = H + (size_t)row * D;
  float acc = 0.0f;
  for (int c = lane; c < D; c += 32) acc += (float)p[c] * w[c];
#pragma unroll
  for (int off = 16; off > 0; off >>= 1) acc += __shfl_xor(acc, off, 32);
  if (lane == 0) out[row] = acc;
}

// ---------------------------------------------------------------------------
// Orchestration
// ---------------------------------------------------------------------------
extern "C" void kernel_launch(void* const* d_in, const int* in_sizes, int n_in,
                              void* d_out, int out_size, void* d_ws, size_t ws_size,
                              hipStream_t stream) {
  const float* x  = (const float*)d_in[0];
  const float* Wq = (const float*)d_in[1];
  const float* bq = (const float*)d_in[2];
  const float* Wk = (const float*)d_in[3];
  const float* bk = (const float*)d_in[4];
  const float* Wv = (const float*)d_in[5];
  const float* bv = (const float*)d_in[6];
  const float* W1 = (const float*)d_in[7];
  const float* b1 = (const float*)d_in[8];
  const float* W2 = (const float*)d_in[9];
  const float* b2 = (const float*)d_in[10];
  const float* W3 = (const float*)d_in[11];
  const float* b3 = (const float*)d_in[12];
  const float* fw = (const float*)d_in[13];
  float* out = (float*)d_out;

  const int D = in_sizes[2];           // bias length = 1024
  const int N = in_sizes[0] / D;       // 8192 tokens

  // ---- workspace carve-out (256B aligned); S (N*N bf16 = 134MB) fits the
  // MI455X's 192MB L2, keeping softmax + attn@V L2-resident.
  char* wsp = (char*)d_ws;
  auto alloc = [&](size_t bytes) -> char* {
    char* p = wsp;
    wsp += (bytes + 255) & ~(size_t)255;
    return p;
  };
  __bf16* xb  = (__bf16*)alloc((size_t)N * D * 2);
  __bf16* Wqb = (__bf16*)alloc((size_t)D * D * 2);
  __bf16* Wkb = (__bf16*)alloc((size_t)D * D * 2);
  __bf16* Wvb = (__bf16*)alloc((size_t)D * D * 2);
  __bf16* W1b = (__bf16*)alloc((size_t)D * D * 2);
  __bf16* W2b = (__bf16*)alloc((size_t)D * D * 2);
  __bf16* W3b = (__bf16*)alloc((size_t)D * D * 2);
  __bf16* Qb  = (__bf16*)alloc((size_t)N * D * 2);
  __bf16* Kb  = (__bf16*)alloc((size_t)N * D * 2);
  __bf16* Vt  = (__bf16*)alloc((size_t)D * N * 2);   // V stored transposed [D, N]
  __bf16* S   = (__bf16*)alloc((size_t)N * N * 2);   // scores / attn, in place
  // buffer reuse (lifetimes are disjoint):
  __bf16* att = Qb;   // attended = P @ V
  __bf16* h1  = Kb;
  __bf16* h2  = xb;
  __bf16* h3  = Vt;

  auto cvt = [&](const float* src, __bf16* dst, size_t n) {
    long n4 = (long)(n / 4);
    cvt_f32_bf16<<<dim3((unsigned)((n4 + 255) / 256)), 256, 0, stream>>>(src, dst, n4);
  };
  cvt(x,  xb,  (size_t)N * D);
  cvt(Wq, Wqb, (size_t)D * D);
  cvt(Wk, Wkb, (size_t)D * D);
  cvt(Wv, Wvb, (size_t)D * D);
  cvt(W1, W1b, (size_t)D * D);
  cvt(W2, W2b, (size_t)D * D);
  cvt(W3, W3b, (size_t)D * D);

  auto gemm = [&](const __bf16* A, const __bf16* B, const float* bias, __bf16* C,
                  int M, int Nn, int K, int ldc, float scale, int relu, int tr) {
    dim3 grid((unsigned)(Nn / 128), (unsigned)(M / 128));
    gemm_bf16_wmma<<<grid, 256, 0, stream>>>(A, B, bias, C, M, Nn, K, ldc,
                                             scale, relu, tr);
  };

  const float sc = 1.0f / sqrtf((float)D);

  gemm(xb, Wqb, bq, Qb, N, D, D, D, 1.0f, 0, 0);          // Q = x @ Wq^T + bq
  gemm(xb, Wkb, bk, Kb, N, D, D, D, 1.0f, 0, 0);          // K = x @ Wk^T + bk
  gemm(xb, Wvb, bv, Vt, N, D, D, N, 1.0f, 0, 1);          // V^T (transposed store)
  gemm(Qb, Kb, nullptr, S, N, N, D, N, sc, 0, 0);         // S = (Q @ K^T) * scale
  softmax_rows<<<dim3((unsigned)N), 256, 0, stream>>>(S, N);
  gemm(S, Vt, nullptr, att, N, D, N, D, 1.0f, 0, 0);      // att = P @ V
  gemm(att, W1b, b1, h1, N, D, D, D, 1.0f, 1, 0);         // relu(att @ W1^T + b1)
  gemm(h1,  W2b, b2, h2, N, D, D, D, 1.0f, 1, 0);
  gemm(h2,  W3b, b3, h3, N, D, D, D, 1.0f, 1, 0);
  final_matvec<<<dim3((unsigned)(N / 8)), 256, 0, stream>>>(h3, fw, out, D);
}
